// aAsynGINLayer_70188355551847
// MI455X (gfx1250) — matrix operand order repaired and checked by reading, not compiled, not run on hardware
//
#include <hip/hip_runtime.h>

// ---------------------------------------------------------------------------
// GIN layer on MI455X (gfx1250, wave32, WMMA).
// Pipeline per call:
//   out  = MLP_lin(xs[0])                       (fused WMMA kernel, acc=0)
//   for k in 0..2:
//     agg  = 0                                  (zero kernel)
//     agg += scatter-add over edges             (warp/edge native f32 atomics)
//     out += MLP_conv_k((1+eps_k)*x_k + agg)    (fused WMMA kernel, acc=1)
//
// Roofline: scatter dominates (~1.5 GB gather+atomic traffic -> ~65 us at
// 23.3 TB/s); the 8 GEMMs are 7.9 GFLOP total and ride on bf16 WMMA with f32
// accumulate, fully fused with BN/ReLU epilogues so the [N,128] intermediates
// never round-trip HBM.
// ---------------------------------------------------------------------------

#define GN 30000
#define GE 480000
#define GD 128
#define GH 3
#define MBLK 64      // rows per workgroup in the MLP kernel
#define STR 136      // LDS row stride (bf16 elems): 136*2B = 272B = 68 dwords
                     // -> 16 consecutive rows land in distinct LDS banks

typedef __bf16 bf16;
typedef __attribute__((ext_vector_type(16))) __bf16 v16bf;
typedef __attribute__((ext_vector_type(8)))  float  v8f;

union FragAB { v16bf v; unsigned u[8]; };
union FragC  { v8f   v; float    f[8]; };

// Non-returning, relaxed, agent-scope f32 atomic add -> native
// global_atomic_add_f32 (no CAS-loop expansion, no pre-op value return).
__device__ __forceinline__ void atomic_add_f32(float* p, float v) {
    (void)__hip_atomic_fetch_add(p, v, __ATOMIC_RELAXED, __HIP_MEMORY_SCOPE_AGENT);
}

// ---------------------------------------------------------------------------
// Zero workspace (graph-capture-safe alternative to hipMemsetAsync)
// ---------------------------------------------------------------------------
__global__ __launch_bounds__(256) void gin_zero_f32(float* __restrict__ p, int n) {
    int i = blockIdx.x * 256 + threadIdx.x;
    if (i < n) p[i] = 0.0f;
}

// ---------------------------------------------------------------------------
// Edge scatter: one wave32 per edge; each lane moves one float4 (4 of 128
// feature floats) with global_atomic_add_f32. Gather side reads 512B
// contiguous per edge (good HBM efficiency, and x rows are L2-resident:
// 15 MB << 192 MB); scatter relies on L2 atomic units.
// ---------------------------------------------------------------------------
__global__ __launch_bounds__(256) void gin_scatter(const float* __restrict__ X,
                                                   const int*   __restrict__ src,
                                                   const int*   __restrict__ dst,
                                                   float*       __restrict__ agg,
                                                   int nE) {
    int lane = threadIdx.x & 31;
    int e    = (blockIdx.x * 256 + threadIdx.x) >> 5;
    if (e >= nE) return;
    int s = src[e];
    int d = dst[e];
    const float4 xv = reinterpret_cast<const float4*>(X + (size_t)s * GD)[lane];
    float* ap = agg + (size_t)d * GD + lane * 4;
    atomic_add_f32(ap + 0, xv.x);
    atomic_add_f32(ap + 1, xv.y);
    atomic_add_f32(ap + 2, xv.z);
    atomic_add_f32(ap + 3, xv.w);
}

// ---------------------------------------------------------------------------
// Fused MLP:  out[rows] (+)= relu(BN(Z @ W1 + b1)) @ W2 + b2
// where Z = X (agg==nullptr) or Z = (1+eps[hop])*X + agg.
// 256 threads = 8 waves; wave w owns a 16x64 tile strip:
//   rows mrow = 16*(w>>1), cols ncol = 64*(w&1); 4 C-tiles of 16x16.
// K-loop: 4 x v_wmma_f32_16x16x32_bf16 per tile.
// ---------------------------------------------------------------------------
__global__ __launch_bounds__(256) void gin_mlp_wmma(
    const float* __restrict__ X,    const float* __restrict__ agg,
    const float* __restrict__ epsv, int hop,
    const float* __restrict__ W1,   const float* __restrict__ b1,
    const float* __restrict__ gam,  const float* __restrict__ bet,
    const float* __restrict__ mu,   const float* __restrict__ var,
    const float* __restrict__ W2,   const float* __restrict__ b2,
    float* __restrict__ out, int accumulate)
{
    __shared__ __align__(16) bf16 Zb[MBLK * STR]; // 17408 B; reused for ReLU output
    __shared__ __align__(16) bf16 Wt[GD * STR];   // 34816 B; W1^T then W2^T (bf16)

    const int tid     = threadIdx.x;
    const int rowBase = blockIdx.x * MBLK;

    // ---- Stage Z: load 64x128 input tile, form GIN update, convert to bf16
    {
        const float4* X4 = reinterpret_cast<const float4*>(X);
        const float4* A4 = agg ? reinterpret_cast<const float4*>(agg) : nullptr;
        const float epsc = agg ? (1.0f + epsv[hop]) : 1.0f;
        #pragma unroll
        for (int i = 0; i < 8; ++i) {
            int idx = tid + i * 256;          // 0..2047 float4 slots
            int r   = idx >> 5;               // 32 float4 per 128-wide row
            int c4  = idx & 31;
            int gr  = rowBase + r;
            float zx = 0.f, zy = 0.f, zz = 0.f, zw = 0.f;
            if (gr < GN) {
                size_t gi = (size_t)gr * 32 + c4;
                float4 xv = X4[gi];
                zx = epsc * xv.x; zy = epsc * xv.y;
                zz = epsc * xv.z; zw = epsc * xv.w;
                if (A4) {
                    float4 av = A4[gi];
                    zx += av.x; zy += av.y; zz += av.z; zw += av.w;
                }
            }
            bf16* zp = &Zb[r * STR + c4 * 4];
            zp[0] = (bf16)zx; zp[1] = (bf16)zy; zp[2] = (bf16)zz; zp[3] = (bf16)zw;
        }
    }
    // ---- Load W1 transposed into LDS as bf16: Wt[n][k] = W1[k][n]
    {
        const float4* W14 = reinterpret_cast<const float4*>(W1);
        #pragma unroll
        for (int i = 0; i < 16; ++i) {
            int idx4 = tid + i * 256;         // 0..4095 float4 slots
            int k    = idx4 >> 5;             // W1 row (K index)
            int n4   = idx4 & 31;
            float4 w = W14[idx4];
            Wt[(4 * n4 + 0) * STR + k] = (bf16)w.x;
            Wt[(4 * n4 + 1) * STR + k] = (bf16)w.y;
            Wt[(4 * n4 + 2) * STR + k] = (bf16)w.z;
            Wt[(4 * n4 + 3) * STR + k] = (bf16)w.w;
        }
    }
    __syncthreads();

    const int lane = tid & 31;
    const int wave = tid >> 5;
    const int mrow = 16 * (wave >> 1);
    const int ncol = 64 * (wave & 1);
    const int half = lane >> 4;
    const int ln   = lane & 15;

    // ---- GEMM1: C = Zb @ W1  (bf16 x bf16 -> f32), K = 128 in 4 steps of 32
    FragC c[4];
    #pragma unroll
    for (int t = 0; t < 4; ++t) c[t].v = v8f{0.f,0.f,0.f,0.f,0.f,0.f,0.f,0.f};

    #pragma unroll
    for (int kb = 0; kb < 4; ++kb) {
        const int K0 = 32 * kb;
        FragAB a;  // ISA 7.12.2: lanes 0-15 row M=ln hold K {0..7,16..23}; +16 lanes hold {8..15,24..31}
        #pragma unroll
        for (int vv = 0; vv < 8; ++vv) {
            int kbase = K0 + 2 * (vv & 3) + 8 * half + 16 * (vv >> 2);
            a.u[vv] = *reinterpret_cast<const unsigned*>(&Zb[(mrow + ln) * STR + kbase]);
        }
        #pragma unroll
        for (int t = 0; t < 4; ++t) {
            const int nc = ncol + 16 * t + ln;
            FragAB b;  // B 32x16: lanes 0-15 K={0..7,16..23}, lanes 16-31 K={8..15,24..31}, col = ln
            #pragma unroll
            for (int vv = 0; vv < 8; ++vv) {
                int kbase = K0 + 2 * (vv & 3) + 8 * half + 16 * (vv >> 2);
                b.u[vv] = *reinterpret_cast<const unsigned*>(&Wt[nc * STR + kbase]);
            }
            c[t].v = __builtin_amdgcn_wmma_f32_16x16x32_bf16(
                false, a.v, false, b.v, (short)0, c[t].v, false, false);
        }
    }
    __syncthreads();   // all GEMM1 reads of Zb/Wt done -> safe to overwrite both

    // ---- Epilogue 1: fused BN (affine) + ReLU -> bf16 back into Zb
    #pragma unroll
    for (int t = 0; t < 4; ++t) {
        const int col = ncol + 16 * t + ln;
        float s    = gam[col] * rsqrtf(var[col] + 1e-5f);
        float bias = (b1[col] - mu[col]) * s + bet[col];
        #pragma unroll
        for (int r = 0; r < 8; ++r) {
            float y = fmaxf(c[t].f[r] * s + bias, 0.0f);
            Zb[(mrow + r + 8 * half) * STR + col] = (bf16)y;
        }
    }
    // ---- Load W2 transposed over Wt
    {
        const float4* W24 = reinterpret_cast<const float4*>(W2);
        #pragma unroll
        for (int i = 0; i < 16; ++i) {
            int idx4 = tid + i * 256;
            int k    = idx4 >> 5;
            int n4   = idx4 & 31;
            float4 w = W24[idx4];
            Wt[(4 * n4 + 0) * STR + k] = (bf16)w.x;
            Wt[(4 * n4 + 1) * STR + k] = (bf16)w.y;
            Wt[(4 * n4 + 2) * STR + k] = (bf16)w.z;
            Wt[(4 * n4 + 3) * STR + k] = (bf16)w.w;
        }
    }
    __syncthreads();

    // ---- GEMM2: C = relu(...) @ W2
    #pragma unroll
    for (int t = 0; t < 4; ++t) c[t].v = v8f{0.f,0.f,0.f,0.f,0.f,0.f,0.f,0.f};

    #pragma unroll
    for (int kb = 0; kb < 4; ++kb) {
        const int K0 = 32 * kb;
        FragAB a;
        #pragma unroll
        for (int vv = 0; vv < 8; ++vv) {
            int kbase = K0 + 2 * (vv & 3) + 8 * half + 16 * (vv >> 2);
            a.u[vv] = *reinterpret_cast<const unsigned*>(&Zb[(mrow + ln) * STR + kbase]);
        }
        #pragma unroll
        for (int t = 0; t < 4; ++t) {
            const int nc = ncol + 16 * t + ln;
            FragAB b;
            #pragma unroll
            for (int vv = 0; vv < 8; ++vv) {
                int kbase = K0 + 2 * (vv & 3) + 8 * half + 16 * (vv >> 2);
                b.u[vv] = *reinterpret_cast<const unsigned*>(&Wt[nc * STR + kbase]);
            }
            c[t].v = __builtin_amdgcn_wmma_f32_16x16x32_bf16(
                false, a.v, false, b.v, (short)0, c[t].v, false, false);
        }
    }

    // ---- Epilogue 2: + b2, write or accumulate into out (f32)
    #pragma unroll
    for (int t = 0; t < 4; ++t) {
        const int col = ncol + 16 * t + ln;
        const float bb = b2[col];
        #pragma unroll
        for (int r = 0; r < 8; ++r) {
            int grow = rowBase + mrow + r + 8 * half;
            if (grow < GN) {
                size_t oi = (size_t)grow * GD + col;
                float y = c[t].f[r] + bb;
                out[oi] = accumulate ? (out[oi] + y) : y;
            }
        }
    }
}

// ---------------------------------------------------------------------------
extern "C" void kernel_launch(void* const* d_in, const int* in_sizes, int n_in,
                              void* d_out, int out_size, void* d_ws, size_t ws_size,
                              hipStream_t stream) {
    const float* xs   = (const float*)d_in[0];   // [H+1, N, D]
    const int*   ei   = (const int*)  d_in[1];   // [H, 2, E]
    const float* lW1  = (const float*)d_in[2];
    const float* lb1  = (const float*)d_in[3];
    const float* lg   = (const float*)d_in[4];
    const float* lbt  = (const float*)d_in[5];
    const float* lm   = (const float*)d_in[6];
    const float* lv   = (const float*)d_in[7];
    const float* lW2  = (const float*)d_in[8];
    const float* lb2  = (const float*)d_in[9];
    const float* epsv = (const float*)d_in[10];  // [H]
    const float* cW1  = (const float*)d_in[11];  // [H, D, D]
    const float* cb1  = (const float*)d_in[12];  // [H, D]
    const float* cg   = (const float*)d_in[13];
    const float* cbt  = (const float*)d_in[14];
    const float* cm   = (const float*)d_in[15];
    const float* cv   = (const float*)d_in[16];
    const float* cW2  = (const float*)d_in[17];
    const float* cb2  = (const float*)d_in[18];
    float* out = (float*)d_out;
    float* agg = (float*)d_ws;                   // N*D f32 scratch = 15.36 MB

    (void)in_sizes; (void)n_in; (void)out_size; (void)ws_size;

    const dim3 blk(256);
    const int mlpGrid  = (GN + MBLK - 1) / MBLK;          // 469
    const int zeroGrid = (GN * GD + 255) / 256;           // 15000
    const int scatGrid = ((GE * 32) + 255) / 256;         // 60000 (one wave32/edge)

    // hidden = lin-MLP(xs[0])
    gin_mlp_wmma<<<mlpGrid, blk, 0, stream>>>(
        xs, nullptr, nullptr, 0,
        lW1, lb1, lg, lbt, lm, lv, lW2, lb2, out, /*accumulate=*/0);

    for (int k = 0; k < GH; ++k) {
        const float* xk   = xs + (size_t)(k + 1) * GN * GD;
        const int*   srck = ei + (size_t)k * 2 * GE;
        const int*   dstk = srck + GE;

        gin_zero_f32<<<zeroGrid, blk, 0, stream>>>(agg, GN * GD);
        gin_scatter <<<scatGrid, blk, 0, stream>>>(xk, srck, dstk, agg, GE);
        gin_mlp_wmma<<<mlpGrid,  blk, 0, stream>>>(
            xk, agg, epsv, k,
            cW1 + (size_t)k * GD * GD, cb1 + (size_t)k * GD,
            cg  + (size_t)k * GD,      cbt + (size_t)k * GD,
            cm  + (size_t)k * GD,      cv  + (size_t)k * GD,
            cW2 + (size_t)k * GD * GD, cb2 + (size_t)k * GD,
            out, /*accumulate=*/1);
    }
}
